// MHA_54236847014292
// MI455X (gfx1250) — matrix operand order repaired and checked by reading, not compiled, not run on hardware
//
#include <hip/hip_runtime.h>

// ---------------------------------------------------------------------------
// MI455X (gfx1250) causal linear attention (Performer-style), chunk-parallel.
//   K1 : fused q/k/v projection GEMMs, bf16 WMMA 16x16x32, 128x128 tiles,
//        global_prefetch of next K-slab.
//   K2 : phi feature map (normalize + dual-row 64x64 matvec + sin/cos).
//   K3a: per-chunk transposed state S_c^T = V^T phi_k  [d][k]  (WMMA, 512 blk)
//   K3b: exclusive prefix scan over 16 chunks -> bf16 [d][k]   (32 blk)
//   K3c: O = phi_q S_prev + mask(phi_q phi_k^T) V, / qz        (WMMA, 512 blk)
//        phi + state tiles staged into LDS via TDM tensor_load_to_lds when
//        available (pad feature produces the 136-half padded stride natively).
// ---------------------------------------------------------------------------

typedef __attribute__((ext_vector_type(16))) __bf16 v16bf;
typedef __attribute__((ext_vector_type(8)))  float  v8f;

#define DEVINL __device__ __forceinline__

DEVINL unsigned short f32_to_bf16(float f) {
  union { float f; unsigned u; } x; x.f = f;
  unsigned r = x.u + 0x7FFFu + ((x.u >> 16) & 1u);   // round-to-nearest-even
  return (unsigned short)(r >> 16);
}
DEVINL float bf16_to_f32(unsigned short h) {
  union { unsigned u; float f; } x; x.u = ((unsigned)h) << 16; return x.f;
}
DEVINL v16bf make_v16bf(uint4 a, uint4 b) {
  union { uint4 u[2]; v16bf v; } c; c.u[0] = a; c.u[1] = b; return c.v;
}

// A fragment (16x32 bf16, M x K): tile stored row-major [row][k], stride `ld`
// halves.  ISA layout: lanes 0-15 -> M rows, K groups {0-7,16-23};
// lanes 16-31 -> same rows, K groups {8-15,24-31}.  ld, kbase multiples of 8.
DEVINL v16bf ld_frag_a(const unsigned short* s, int ld, int rbase, int kbase) {
  int l = (int)(threadIdx.x & 31u);
  int hs = l >> 4;
  const unsigned short* p = s + (rbase + (l & 15)) * ld + kbase;
  uint4 x = *(const uint4*)(p + (hs ? 8 : 0));
  uint4 y = *(const uint4*)(p + (hs ? 24 : 16));
  return make_v16bf(x, y);
}
// B fragment (32x16 bf16, K x N): tile stored N-major Bt[n][k], stride `ld`.
DEVINL v16bf ld_frag_b(const unsigned short* s, int ld, int nbase, int kbase) {
  int l = (int)(threadIdx.x & 31u);
  int hs = l >> 4;
  const unsigned short* p = s + (nbase + (l & 15)) * ld + kbase;
  uint4 x = *(const uint4*)(p + (hs ? 8 : 0));
  uint4 y = *(const uint4*)(p + (hs ? 24 : 16));
  return make_v16bf(x, y);
}

DEVINL v8f wmma_bf16(v16bf a, v16bf b, v8f c) {
  return __builtin_amdgcn_wmma_f32_16x16x32_bf16(
      /*neg_a=*/false, a, /*neg_b=*/false, b,
      /*c_mod=*/(short)0, c, /*reuse_a=*/false, /*reuse_b=*/false);
}

// ---------------------------------------------------------------------------
// TDM: 2D bf16 tile load, 64 rows x 128 halves, parameterized memory row
// stride; LDS padded 4 DWORDs after every 64 DWORDs (-> 136-half row stride).
// Descriptor per CDNA5 ISA §8.3 (group0) / §8.4 (group1).
// ---------------------------------------------------------------------------
#if __has_builtin(__builtin_amdgcn_tensor_load_to_lds) && __has_builtin(__builtin_amdgcn_s_wait_tensorcnt)
#define HAVE_TDM 1
typedef __attribute__((ext_vector_type(4))) unsigned int tdm_v4u;
typedef __attribute__((ext_vector_type(4))) int          tdm_v4i;
typedef __attribute__((ext_vector_type(8))) int          tdm_v8i;

DEVINL unsigned lds_byte_off(const void* p) {
  // generic LDS pointer: addr[31:0] is the workgroup-relative LDS byte address
  return (unsigned)(unsigned long long)p;
}

DEVINL void tdm_load_tile_64x128(const unsigned short* gsrc, unsigned lds_off,
                                 int row_stride_elems) {
  unsigned long long ga = (unsigned long long)gsrc;
  tdm_v4u g0;
  g0[0] = 1u;                                        // count=1 (valid user D#)
  g0[1] = lds_off;                                   // lds_addr (bytes)
  g0[2] = (unsigned)(ga & 0xFFFFFFFFu);              // global_addr[31:0]
  g0[3] = (unsigned)((ga >> 32) & 0x01FFFFFFu)       // global_addr[56:32]
        | (2u << 30);                                // type = 2 ("image")
  tdm_v8i g1;
  g1[0] = (int)((1u << 16)        // data_size = 1 -> 2 bytes
              | (1u << 20)        // pad_enable
              | (5u << 22)        // pad_interval: 64 DWORDs
              | (3u << 25));      // pad_amount:   4 DWORDs
  g1[1] = (int)(128u << 16);      // tensor_dim0 = 128 elems (bits 79:48 lo)
  g1[2] = (int)(64u << 16);       // tensor_dim1 = 64        (bits 111:80 lo)
  g1[3] = (int)(128u << 16);      // tile_dim0 = 128         (bits 127:112)
  g1[4] = (int)(64u);             // tile_dim1 = 64, tile_dim2 = 0
  g1[5] = row_stride_elems;       // tensor_dim0_stride (lo 32)
  g1[6] = 0;
  g1[7] = 0;
  tdm_v4i z4 = {0, 0, 0, 0};
#if defined(__clang_major__) && (__clang_major__ >= 23)
  tdm_v8i z8 = {0, 0, 0, 0, 0, 0, 0, 0};
  __builtin_amdgcn_tensor_load_to_lds(g0, g1, z4, z4, z8, 0);
#else
  __builtin_amdgcn_tensor_load_to_lds(g0, g1, z4, z4, 0);
#endif
}
#endif  // HAVE_TDM

// ---------------------------------------------------------------------------
// K1: projection GEMM.  out[M=2048,N=1024] = X[M,K=1024] * W[K,N] + b[N]
// ---------------------------------------------------------------------------
#define LDA 72   // padded LDS stride in halves (multiple of 8)

__global__ __launch_bounds__(256) void proj_gemm_kernel(
    const float* __restrict__ q, const float* __restrict__ k, const float* __restrict__ v,
    const float* __restrict__ Wq, const float* __restrict__ Wk, const float* __restrict__ Wv,
    const float* __restrict__ bq, const float* __restrict__ bk, const float* __restrict__ bv,
    float* __restrict__ qp, float* __restrict__ kp, float* __restrict__ vp) {
  __shared__ unsigned short Al[128 * LDA];   // A tile, row-major [m][k]
  __shared__ unsigned short Bl[128 * LDA];   // B tile, N-major  [n][k]

  const float* X; const float* W; const float* bias; float* Out;
  if (blockIdx.z == 0)      { X = q; W = Wq; bias = bq; Out = qp; }
  else if (blockIdx.z == 1) { X = k; W = Wk; bias = bk; Out = kp; }
  else                      { X = v; W = Wv; bias = bv; Out = vp; }

  const int tid = threadIdx.x;
  const int m0 = blockIdx.y * 128, n0 = blockIdx.x * 128;
  const int w = tid >> 5, l = tid & 31, hs = l >> 4, ln = l & 15;
  const int wr = w >> 2, wc = w & 3;     // 2 x 4 wave grid

  const v8f vzero = {0.f,0.f,0.f,0.f,0.f,0.f,0.f,0.f};
  v8f acc[4][2];
  for (int i = 0; i < 4; i++) for (int j = 0; j < 2; j++) acc[i][j] = vzero;

  for (int kk = 0; kk < 1024; kk += 64) {
    // prefetch next K-slab (lowers to global_prefetch_b8)
    if (kk + 64 < 1024) {
      int prow = tid >> 1;            // 128 rows, 2 threads each -> 2x128B lines
      __builtin_prefetch(X + (size_t)(m0 + prow) * 1024 + kk + 64 + (tid & 1) * 32, 0, 1);
      int pkr = tid >> 2;             // 64 K-rows, 4 threads each
      __builtin_prefetch(W + (size_t)(kk + 64 + pkr) * 1024 + n0 + (tid & 3) * 32, 0, 1);
    }
    // stage A: 128 rows x 64 K, fp32 -> bf16
    for (int c = tid; c < 128 * 16; c += 256) {
      int row = c >> 4, kc = (c & 15) * 4;
      float4 f = *(const float4*)(X + (size_t)(m0 + row) * 1024 + kk + kc);
      unsigned short* d = Al + row * LDA + kc;
      d[0] = f32_to_bf16(f.x); d[1] = f32_to_bf16(f.y);
      d[2] = f32_to_bf16(f.z); d[3] = f32_to_bf16(f.w);
    }
    // stage B transposed: K rows kk..kk+63, N cols n0..n0+127 -> Bt[n][k]
    for (int c = tid; c < 64 * 32; c += 256) {
      int kr = c >> 5, nc = (c & 31) * 4;
      float4 f = *(const float4*)(W + (size_t)(kk + kr) * 1024 + n0 + nc);
      Bl[(nc + 0) * LDA + kr] = f32_to_bf16(f.x);
      Bl[(nc + 1) * LDA + kr] = f32_to_bf16(f.y);
      Bl[(nc + 2) * LDA + kr] = f32_to_bf16(f.z);
      Bl[(nc + 3) * LDA + kr] = f32_to_bf16(f.w);
    }
    __syncthreads();

    #pragma unroll
    for (int ks = 0; ks < 64; ks += 32) {
      v16bf af[4], bfv[2];
      #pragma unroll
      for (int i = 0; i < 4; i++) af[i] = ld_frag_a(Al, LDA, wr * 64 + i * 16, ks);
      #pragma unroll
      for (int j = 0; j < 2; j++) bfv[j] = ld_frag_b(Bl, LDA, wc * 32 + j * 16, ks);
      #pragma unroll
      for (int i = 0; i < 4; i++)
        #pragma unroll
        for (int j = 0; j < 2; j++)
          acc[i][j] = wmma_bf16(af[i], bfv[j], acc[i][j]);
    }
    __syncthreads();
  }

  for (int i = 0; i < 4; i++) {
    for (int j = 0; j < 2; j++) {
      int nn = n0 + wc * 32 + j * 16 + ln;
      float bias_n = bias[nn];
      #pragma unroll
      for (int r = 0; r < 8; r++) {
        int mm = m0 + wr * 64 + i * 16 + hs * 8 + r;
        Out[(size_t)mm * 1024 + nn] = acc[i][j][r] + bias_n;
      }
    }
  }
}

// ---------------------------------------------------------------------------
// K2: phi feature map: xn = x/(||x||+1e-3); p = xn @ rm_h^T;
//     phi = 0.1*[sin(p),cos(p)] -> bf16 [2048,16,128].  rm = sigma*rand_mat.
// Two projection rows per pass (shares xr reads); packed b32 stores.
// ---------------------------------------------------------------------------
__global__ __launch_bounds__(256) void phi_kernel(
    const float* __restrict__ qp, const float* __restrict__ kp,
    const float* __restrict__ sigma, const float* __restrict__ rmat,
    unsigned short* __restrict__ phiq, unsigned short* __restrict__ phik) {
  __shared__ float rm[64 * 64];
  const int h = blockIdx.y;
  const int tid = threadIdx.x;
  for (int c = tid; c < 4096; c += 256)
    rm[c] = sigma[h * 64 + (c & 63)] * rmat[(size_t)h * 4096 + c];
  __syncthreads();

  const float* xp = (blockIdx.z == 0) ? qp : kp;
  unsigned short* op = (blockIdx.z == 0) ? phiq : phik;

  const int row = blockIdx.x * 256 + tid;          // over B*S = 2048
  const float* x = xp + (size_t)row * 1024 + h * 64;

  float xr[64];
  float nrm = 0.f;
  #pragma unroll
  for (int d = 0; d < 64; d++) { xr[d] = x[d]; nrm += xr[d] * xr[d]; }
  float inv = 1.f / (sqrtf(nrm) + 0.001f);
  #pragma unroll
  for (int d = 0; d < 64; d++) xr[d] *= inv;

  unsigned short* o = op + ((size_t)row * 16 + h) * 128;
  for (int k2 = 0; k2 < 64; k2 += 2) {
    float p0 = 0.f, p1 = 0.f;
    #pragma unroll
    for (int d = 0; d < 64; d++) {
      float xv = xr[d];
      p0 += xv * rm[k2 * 64 + d];
      p1 += xv * rm[(k2 + 1) * 64 + d];
    }
    unsigned s01 = (unsigned)f32_to_bf16(__sinf(p0) * 0.1f)
                 | ((unsigned)f32_to_bf16(__sinf(p1) * 0.1f) << 16);
    unsigned c01 = (unsigned)f32_to_bf16(__cosf(p0) * 0.1f)
                 | ((unsigned)f32_to_bf16(__cosf(p1) * 0.1f) << 16);
    *(unsigned*)(o + k2)      = s01;
    *(unsigned*)(o + 64 + k2) = c01;
  }
}

// ---------------------------------------------------------------------------
// K3a: per-chunk TRANSPOSED state.  grid (16 chunks, 32 bh).
//   Schunk[bh][c][64 d][128 k] = (phi_k^T V)^T = V^T phi_k  (WMMA: A=Vt, B=PkT)
//   Zchunk[bh][c][128]         = colsum(phi_k)
// Transposed form makes the D-fragment stores coalesced and lets K3b/K3c
// stream it contiguously.
// ---------------------------------------------------------------------------
__global__ __launch_bounds__(256) void chunk_state_kernel(
    const unsigned short* __restrict__ phik, const float* __restrict__ vp,
    float* __restrict__ Schunk, float* __restrict__ Zchunk) {
  __shared__ unsigned short PkT[128 * 72];   // [k][s]
  __shared__ unsigned short Vt[64 * 72];     // [d][s]
  const int chunk = blockIdx.x, bh = blockIdx.y;
  const int b = bh >> 4, h = bh & 15;
  const int s0 = chunk * 64;
  const int tid = threadIdx.x;
  const int w = tid >> 5, l = tid & 31, hs = l >> 4, ln = l & 15;
  const v8f vzero = {0.f,0.f,0.f,0.f,0.f,0.f,0.f,0.f};

  const size_t gbase = ((size_t)(b * 1024 + s0) * 16 + h) * 128;
  // load phi_k rows as uint4 (8 halves) and scatter transposed into PkT
  for (int c = tid; c < 64 * 16; c += 256) {
    int si = c >> 4, kc = (c & 15) * 8;
    uint4 kv = *(const uint4*)(phik + gbase + (size_t)si * 2048 + kc);
    PkT[(kc + 0) * 72 + si] = (unsigned short)(kv.x & 0xffffu);
    PkT[(kc + 1) * 72 + si] = (unsigned short)(kv.x >> 16);
    PkT[(kc + 2) * 72 + si] = (unsigned short)(kv.y & 0xffffu);
    PkT[(kc + 3) * 72 + si] = (unsigned short)(kv.y >> 16);
    PkT[(kc + 4) * 72 + si] = (unsigned short)(kv.z & 0xffffu);
    PkT[(kc + 5) * 72 + si] = (unsigned short)(kv.z >> 16);
    PkT[(kc + 6) * 72 + si] = (unsigned short)(kv.w & 0xffffu);
    PkT[(kc + 7) * 72 + si] = (unsigned short)(kv.w >> 16);
  }
  for (int c = tid; c < 64 * 64; c += 256) {
    int si = c >> 6, d = c & 63;
    Vt[d * 72 + si] = f32_to_bf16(vp[(size_t)(b * 1024 + s0 + si) * 1024 + h * 64 + d]);
  }
  __syncthreads();

  // S^T[d][k]: M = d (4 tiles), N = kdim (8 tiles), K = s (64)
  float* SoutT = Schunk + ((size_t)bh * 16 + chunk) * 8192;
  const int mt = w >> 1;             // d tile 0..3
  const int ntb = (w & 1) * 4;       // kdim tiles ntb..ntb+3
  for (int nt = ntb; nt < ntb + 4; nt++) {
    v8f acc = vzero;
    #pragma unroll
    for (int ks = 0; ks < 64; ks += 32)
      acc = wmma_bf16(ld_frag_a(Vt, 72, mt * 16, ks),
                      ld_frag_b(PkT, 72, nt * 16, ks), acc);
    #pragma unroll
    for (int r = 0; r < 8; r++)
      SoutT[(mt * 16 + hs * 8 + r) * 128 + nt * 16 + ln] = acc[r];
  }
  if (tid < 128) {
    float acc = 0.f;
    for (int si = 0; si < 64; si++) acc += bf16_to_f32(PkT[tid * 72 + si]);
    Zchunk[((size_t)bh * 16 + chunk) * 128 + tid] = acc;
  }
}

// ---------------------------------------------------------------------------
// K3b: exclusive prefix scan along the 16 chunks, per bh.  fp32 in, bf16 out
// (transposed [d][k] layout is contiguous, so both sides stream coalesced).
// ---------------------------------------------------------------------------
__global__ __launch_bounds__(256) void state_scan_kernel(
    const float* __restrict__ Schunk, unsigned short* __restrict__ SbT,
    float* __restrict__ Zchunk) {
  const int bh = blockIdx.x;
  const float* S = Schunk + (size_t)bh * 16 * 8192;
  unsigned short* T = SbT + (size_t)bh * 16 * 8192;
  for (int idx = threadIdx.x; idx < 8192; idx += 256) {
    float run = 0.f;
    #pragma unroll
    for (int c = 0; c < 16; c++) {
      float t = S[c * 8192 + idx];
      T[c * 8192 + idx] = f32_to_bf16(run);   // exclusive prefix, bf16
      run += t;
    }
  }
  if (threadIdx.x < 128) {
    float* Z = Zchunk + (size_t)bh * 16 * 128;
    float run = 0.f;
    #pragma unroll
    for (int c = 0; c < 16; c++) {
      float t = Z[c * 128 + threadIdx.x];
      Z[c * 128 + threadIdx.x] = run;
      run += t;
    }
  }
}

// ---------------------------------------------------------------------------
// K3c: per-chunk output.  grid (16 chunks, 32 bh).
//   A   = mask_lower(Pq @ Pk^T)          [64x64], K=128 (WMMA)
//   O   = Pq @ S_prev + A @ V            [64x64], K=128/64 (WMMA)
//   qz  = Pq . z_prev + rowsum(A), clipped to [1, 1e10]
//   out = O / qz    (flat layout ((b*H+h)*S + s)*D + d per reference reshape)
// Pq / Pk / state tiles staged by the Tensor Data Mover when available.
// ---------------------------------------------------------------------------
__global__ __launch_bounds__(256) void attn_out_kernel(
    const unsigned short* __restrict__ phiq, const unsigned short* __restrict__ phik,
    const float* __restrict__ vp, const unsigned short* __restrict__ SbT,
    const float* __restrict__ Zchunk, float* __restrict__ out) {
  __shared__ unsigned short StBt[64 * 136];  // state prefix, N-major [d][k]
  __shared__ unsigned short Pq[64 * 136];    // [s][k]
  __shared__ unsigned short Pk[64 * 136];    // [s][k]
  __shared__ unsigned short Vt[64 * 72];     // [d][s]
  __shared__ unsigned short Am[64 * 72];     // masked A, [i][j]
  __shared__ float zprev[128];
  __shared__ float qzv[64];

  const int chunk = blockIdx.x, bh = blockIdx.y;
  const int b = bh >> 4, h = bh & 15;
  const int s0 = chunk * 64;
  const int tid = threadIdx.x;
  const int w = tid >> 5, l = tid & 31, hs = l >> 4, ln = l & 15;
  const v8f vzero = {0.f,0.f,0.f,0.f,0.f,0.f,0.f,0.f};

  const size_t gbase = ((size_t)(b * 1024 + s0) * 16 + h) * 128;
  const unsigned short* Stile = SbT + ((size_t)bh * 16 + chunk) * 8192;
#ifdef HAVE_TDM
  if (w == 0)      tdm_load_tile_64x128(phiq + gbase, lds_byte_off(Pq), 2048);
  else if (w == 1) tdm_load_tile_64x128(phik + gbase, lds_byte_off(Pk), 2048);
  else if (w == 2) tdm_load_tile_64x128(Stile, lds_byte_off(StBt), 128);
#else
  for (int c = tid; c < 64 * 16; c += 256) {
    int si = c >> 4, kc = (c & 15) * 8;
    size_t g = gbase + (size_t)si * 2048 + kc;
    *(uint4*)(Pq + si * 136 + kc) = *(const uint4*)(phiq + g);
    *(uint4*)(Pk + si * 136 + kc) = *(const uint4*)(phik + g);
    *(uint4*)(StBt + si * 136 + kc) = *(const uint4*)(Stile + si * 128 + kc);
  }
#endif
  for (int c = tid; c < 64 * 64; c += 256) {
    int si = c >> 6, d = c & 63;
    Vt[d * 72 + si] = f32_to_bf16(vp[(size_t)(b * 1024 + s0 + si) * 1024 + h * 64 + d]);
  }
  if (tid < 128) zprev[tid] = Zchunk[((size_t)bh * 16 + chunk) * 128 + tid];
#ifdef HAVE_TDM
  if (w < 3) __builtin_amdgcn_s_wait_tensorcnt(0);
#endif
  __syncthreads();

  const int mt = w >> 1, nt0 = (w & 1) * 2, nt1 = nt0 + 1;

  // ---- A_intra = mask(Pq @ Pk^T) ----
  {
    v8f a0 = vzero, a1 = vzero;
    #pragma unroll
    for (int ks = 0; ks < 128; ks += 32) {
      v16bf af = ld_frag_a(Pq, 136, mt * 16, ks);
      a0 = wmma_bf16(af, ld_frag_b(Pk, 136, nt0 * 16, ks), a0);
      a1 = wmma_bf16(af, ld_frag_b(Pk, 136, nt1 * 16, ks), a1);
    }
    #pragma unroll
    for (int r = 0; r < 8; r++) {
      int i = mt * 16 + hs * 8 + r;
      int j0 = nt0 * 16 + ln, j1 = nt1 * 16 + ln;
      Am[i * 72 + j0] = f32_to_bf16(j0 <= i ? a0[r] : 0.f);
      Am[i * 72 + j1] = f32_to_bf16(j1 <= i ? a1[r] : 0.f);
    }
  }
  __syncthreads();

  // ---- qz ----
  if (tid < 64) {
    float acc = 0.f;
    for (int kd = 0; kd < 128; kd++) acc += bf16_to_f32(Pq[tid * 136 + kd]) * zprev[kd];
    for (int j = 0; j < 64; j++) acc += bf16_to_f32(Am[tid * 72 + j]);
    qzv[tid] = fminf(fmaxf(acc, 1.0f), 1.0e10f);
  }

  // ---- O = Pq @ S_prev (K=128) + Am @ V (K=64) ----
  v8f o0 = vzero, o1 = vzero;
  #pragma unroll
  for (int ks = 0; ks < 128; ks += 32) {
    v16bf af = ld_frag_a(Pq, 136, mt * 16, ks);
    o0 = wmma_bf16(af, ld_frag_b(StBt, 136, nt0 * 16, ks), o0);
    o1 = wmma_bf16(af, ld_frag_b(StBt, 136, nt1 * 16, ks), o1);
  }
  #pragma unroll
  for (int ks = 0; ks < 64; ks += 32) {
    v16bf af = ld_frag_a(Am, 72, mt * 16, ks);
    o0 = wmma_bf16(af, ld_frag_b(Vt, 72, nt0 * 16, ks), o0);
    o1 = wmma_bf16(af, ld_frag_b(Vt, 72, nt1 * 16, ks), o1);
  }
  __syncthreads();     // qzv ready

  #pragma unroll
  for (int r = 0; r < 8; r++) {
    int i = mt * 16 + hs * 8 + r;
    float inv_qz = 1.f / qzv[i];
    size_t gr = ((size_t)bh * 1024 + (s0 + i)) * 64;
    out[gr + nt0 * 16 + ln] = o0[r] * inv_qz;
    out[gr + nt1 * 16 + ln] = o1[r] * inv_qz;
  }
}

// ---------------------------------------------------------------------------
extern "C" void kernel_launch(void* const* d_in, const int* in_sizes, int n_in,
                              void* d_out, int out_size, void* d_ws, size_t ws_size,
                              hipStream_t stream) {
  (void)in_sizes; (void)n_in; (void)out_size; (void)ws_size;
  const float* query = (const float*)d_in[0];
  const float* key   = (const float*)d_in[1];
  const float* value = (const float*)d_in[2];
  const float* Wq    = (const float*)d_in[3];
  const float* bq    = (const float*)d_in[4];
  const float* Wk    = (const float*)d_in[5];
  const float* bk    = (const float*)d_in[6];
  const float* Wv    = (const float*)d_in[7];
  const float* bv    = (const float*)d_in[8];
  const float* sigma = (const float*)d_in[9];
  const float* rmat  = (const float*)d_in[10];
  float* out = (float*)d_out;

  // workspace carve:
  //   qp/kp/vp fp32 [2048,1024]            3 x 8 MB
  //   phi_q/phi_k bf16 [2048,16,128]       2 x 8 MB
  //   Schunk fp32 [32][16][64][128]        16 MB   (per-chunk S^T sums)
  //   Zchunk fp32 [32][16][128]            256 KB
  //   SbT    bf16 [32][16][64][128]        8 MB    (exclusive prefix, bf16)
  float* qp = (float*)d_ws;
  float* kp = qp + (size_t)2048 * 1024;
  float* vp = kp + (size_t)2048 * 1024;
  unsigned short* phiq = (unsigned short*)(vp + (size_t)2048 * 1024);
  unsigned short* phik = phiq + (size_t)2048 * 16 * 128;
  float* Schunk = (float*)(phik + (size_t)2048 * 16 * 128);
  float* Zchunk = Schunk + (size_t)32 * 16 * 8192;
  unsigned short* SbT = (unsigned short*)(Zchunk + (size_t)32 * 16 * 128);

  proj_gemm_kernel<<<dim3(8, 16, 3), 256, 0, stream>>>(
      query, key, value, Wq, Wk, Wv, bq, bk, bv, qp, kp, vp);
  phi_kernel<<<dim3(8, 16, 2), 256, 0, stream>>>(qp, kp, sigma, rmat, phiq, phik);
  chunk_state_kernel<<<dim3(16, 32), 256, 0, stream>>>(phik, vp, Schunk, Zchunk);
  state_scan_kernel<<<dim3(32), 256, 0, stream>>>(Schunk, SbT, Zchunk);
  attn_out_kernel<<<dim3(16, 32), 256, 0, stream>>>(phiq, phik, vp, SbT, Zchunk, out);
}
